// GNN_lrelu_50689204027573
// MI455X (gfx1250) — compile-verified, matrix operand order checked by reading, not run on hardware
//
#include <hip/hip_runtime.h>

typedef __bf16 bf16_t;
typedef __attribute__((ext_vector_type(16))) __bf16 v16bf;
typedef __attribute__((ext_vector_type(8)))  __bf16 v8bf;
typedef __attribute__((ext_vector_type(8)))  float  v8f;

#define N_PFAS 50000
#define N_GW   100000
#define N_SW   20000
#define D      256
#define LDK    40   // padded LDS k-stride (bf16): 80B rows keep 16B chunks aligned, spreads banks
#define MT     64   // M tile per block

// ---------------- zero workspace (agg + counts) ----------------
__global__ void zero4_kernel(float4* __restrict__ p, int n4) {
    int i = blockIdx.x * 256 + threadIdx.x;
    if (i < n4) p[i] = make_float4(0.f, 0.f, 0.f, 0.f);
}

// ------- fp32 weights -> bf16, TRANSPOSED (out[n*256+k] = W[k*256+n] (+W2)) -------
__global__ void wconv_t_kernel(const float* __restrict__ W, const float* __restrict__ W2,
                               bf16_t* __restrict__ out) {
    int i = blockIdx.x * 256 + threadIdx.x;   // 65536 elements; i = k*256 + n
    int k = i >> 8, n = i & 255;
    float v = W[i];
    if (W2) v += W2[i];
    out[n * 256 + k] = (bf16_t)v;
}

// ---------------- edge scatter-add: agg[dst] += x[src], cnt[dst] += 1 ----------------
__global__ void scatter_kernel(const float* __restrict__ x,
                               const int* __restrict__ src,
                               const int* __restrict__ dst,
                               float* __restrict__ agg, float* __restrict__ cnt, int E) {
    int e = blockIdx.x * 4 + (threadIdx.x >> 6);
    if (e >= E) return;
    int l = threadIdx.x & 63;
    int s = src[e], d = dst[e];
    const float* xs = x + (size_t)s * D;
    float* ag = agg + (size_t)d * D;
#pragma unroll
    for (int c = l; c < D; c += 64) atomicAdd(&ag[c], xs[c]);   // global_atomic_add_f32
    if (l == 0) atomicAdd(&cnt[d], 1.0f);
}

// ---- fused SAGE GEMM: sum_mat A_mat @ B_mat (+bias) -> relu -> [head | store] ----
// A fp32 (edge-sum/cnt -> mean if cnt!=null, else raw features); BT bf16 pre-transposed [N][K].
// Block = 256 thr = 8 waves; tile = 64 rows x 256 cols.
// wave w: m_sub = w>>1 (0..3), nb = w&1; frag t=0..7 -> n-tile nt = nb + 2t.
__global__ __launch_bounds__(256)
void gemm_kernel(const float* A0, const float* A1, const float* A2,
                 const bf16_t* B0, const bf16_t* B1, const bf16_t* B2,
                 const float* cnt0, const float* cnt1, const float* cnt2,
                 int nmat,
                 const float* __restrict__ bias0, const float* __restrict__ bias1,
                 const float* __restrict__ headW, const float* __restrict__ headB,
                 float* __restrict__ out, int nrows, int head_mode) {
    __shared__ alignas(16) bf16_t As[MT * LDK];
    __shared__ alignas(16) bf16_t Bs[256 * LDK];
    __shared__ float rowinv[MT];
    __shared__ float rowacc[MT];

    const int tid   = threadIdx.x;
    const int lane  = tid & 31;
    const int w     = tid >> 5;
    const int m0    = blockIdx.x * MT;
    const int m_sub = w >> 1;
    const int nb    = w & 1;
    const int hi    = (lane >= 16) ? 1 : 0;

    if (head_mode && tid < MT) rowacc[tid] = 0.0f;

    v8f acc[8];
#pragma unroll
    for (int t = 0; t < 8; ++t)
#pragma unroll
        for (int i = 0; i < 8; ++i) acc[t][i] = 0.0f;

    for (int mat = 0; mat < nmat; ++mat) {
        const float*  A   = (mat == 0) ? A0   : ((mat == 1) ? A1   : A2);
        const bf16_t* BT  = (mat == 0) ? B0   : ((mat == 1) ? B1   : B2);
        const float*  cnt = (mat == 0) ? cnt0 : ((mat == 1) ? cnt1 : cnt2);

        if (tid < MT) {
            float iv = 1.0f;
            if (cnt) {
                int r = m0 + tid; if (r >= nrows) r = nrows - 1;
                iv = 1.0f / fmaxf(cnt[r], 1.0f);     // SAGE mean divisor
            }
            rowinv[tid] = iv;
        }
        __syncthreads();

        for (int k0 = 0; k0 < D; k0 += 32) {
            // ---- stage A tile (64 rows x 32 k): 8 fp32 per thread -> bf16, scaled ----
            {
                int r  = tid >> 2;          // 0..63
                int kk = (tid & 3) * 8;     // 0, 8, 16, 24
                float4 v0 = make_float4(0.f, 0.f, 0.f, 0.f), v1 = v0;
                int row = m0 + r;
                if (row < nrows) {
                    const float* ap = A + (size_t)row * D + k0 + kk;
                    v0 = *reinterpret_cast<const float4*>(ap);
                    v1 = *reinterpret_cast<const float4*>(ap + 4);
                    if (k0 + 32 < D)
                        __builtin_prefetch(ap + 32, 0, 1);   // global_prefetch_b8: next K tile
                }
                float s = rowinv[r];
                bf16_t* p = As + r * LDK + kk;
                p[0] = (bf16_t)(v0.x * s); p[1] = (bf16_t)(v0.y * s);
                p[2] = (bf16_t)(v0.z * s); p[3] = (bf16_t)(v0.w * s);
                p[4] = (bf16_t)(v1.x * s); p[5] = (bf16_t)(v1.y * s);
                p[6] = (bf16_t)(v1.z * s); p[7] = (bf16_t)(v1.w * s);
            }
            // ---- stage B tile from pre-transposed BT[N][K]: vector copy, 4x b128 each ----
            {
                int c = tid & 3;            // k-chunk of 8
                int n = tid >> 2;           // 0..63, 4 passes of 64 rows
#pragma unroll
                for (int pass = 0; pass < 4; ++pass, n += 64) {
                    v8bf v = *reinterpret_cast<const v8bf*>(BT + (size_t)n * D + k0 + c * 8);
                    *reinterpret_cast<v8bf*>(Bs + n * LDK + c * 8) = v;
                }
            }
            __syncthreads();

            // ---- fragments + WMMA (8 per wave per K-step) ----
            v16bf a;
            {
                const v8bf* pa = reinterpret_cast<const v8bf*>(
                    As + (m_sub * 16 + (lane & 15)) * LDK + hi * 8);
                v8bf lo = pa[0], hi8 = pa[2];   // k-chunks {0..7|8..15}, {16..23|24..31}
#pragma unroll
                for (int i = 0; i < 8; ++i) { a[i] = lo[i]; a[8 + i] = hi8[i]; }
            }
#pragma unroll
            for (int t = 0; t < 8; ++t) {
                int nt = nb + 2 * t;
                const v8bf* pb = reinterpret_cast<const v8bf*>(
                    Bs + (nt * 16 + (lane & 15)) * LDK + hi * 8);
                v8bf lo = pb[0], hi8 = pb[2];
                v16bf b;
#pragma unroll
                for (int i = 0; i < 8; ++i) { b[i] = lo[i]; b[8 + i] = hi8[i]; }
                acc[t] = __builtin_amdgcn_wmma_f32_16x16x32_bf16(
                    false, a, false, b, (short)0, acc[t], false, false);
            }
            __syncthreads();
        }
    }

    // ---- epilogue: bias + relu, then head-dot (ds_add_f32 reduce) or store h ----
#pragma unroll
    for (int t = 0; t < 8; ++t) {
        int col = (nb + 2 * t) * 16 + (lane & 15);
        float bsum = bias0[col];
        if (bias1) bsum += bias1[col];
        float hw = head_mode ? headW[col] : 0.0f;
#pragma unroll
        for (int i = 0; i < 8; ++i) {
            int lrow = m_sub * 16 + i + (hi ? 8 : 0);   // C layout: VGPR i -> M=i / i+8
            float h = acc[t][i] + bsum;
            h = fmaxf(h, 0.0f);                          // relu
            if (head_mode) {
                atomicAdd(&rowacc[lrow], h * hw);        // ds_add_f32
            } else {
                int row = m0 + lrow;
                if (row < nrows) out[(size_t)row * D + col] = h;
            }
        }
    }
    if (head_mode) {
        __syncthreads();
        if (tid < MT) {
            int row = m0 + tid;
            if (row < nrows) {
                float v = rowacc[tid] + headB[0];
                out[row] = (v > 0.0f) ? v : 0.001f * v;  // leaky_relu(0.001)
            }
        }
    }
}

extern "C" void kernel_launch(void* const* d_in, const int* in_sizes, int n_in,
                              void* d_out, int out_size, void* d_ws, size_t ws_size,
                              hipStream_t stream) {
    (void)n_in; (void)out_size; (void)ws_size;
    const float* x_pfas = (const float*)d_in[0];
    const float* x_gw   = (const float*)d_in[1];
    const float* x_sw   = (const float*)d_in[2];
    const int* e_pg_src = (const int*)d_in[3];
    const int* e_pg_dst = (const int*)d_in[4];
    const int* e_gp_src = (const int*)d_in[5];
    const int* e_gp_dst = (const int*)d_in[6];
    const int* e_ps_src = (const int*)d_in[7];
    const int* e_ps_dst = (const int*)d_in[8];
    const int* e_sp_src = (const int*)d_in[9];
    const int* e_sp_dst = (const int*)d_in[10];
    const float* Wl_pg = (const float*)d_in[11]; const float* bl_pg = (const float*)d_in[12]; const float* Wr_pg = (const float*)d_in[13];
    const float* Wl_gp = (const float*)d_in[14]; const float* bl_gp = (const float*)d_in[15]; const float* Wr_gp = (const float*)d_in[16];
    const float* Wl_ps = (const float*)d_in[17]; const float* bl_ps = (const float*)d_in[18]; const float* Wr_ps = (const float*)d_in[19];
    const float* Wl_sp = (const float*)d_in[20]; const float* bl_sp = (const float*)d_in[21]; const float* Wr_sp = (const float*)d_in[22];
    const float* W_gw    = (const float*)d_in[23]; const float* b_gw    = (const float*)d_in[24];
    const float* W_sites = (const float*)d_in[25]; const float* b_sites = (const float*)d_in[26];
    const int E = in_sizes[3];
    float* outp = (float*)d_out;

    // workspace: fp32 edge-sum accumulators + counts, then bf16 transposed weights
    float* agg_pg = (float*)d_ws;                        // [N_GW,  D]
    float* agg_gp = agg_pg + (size_t)N_GW  * D;          // [N_PFAS,D]
    float* agg_ps = agg_gp + (size_t)N_PFAS * D;         // [N_SW,  D]
    float* agg_sp = agg_ps + (size_t)N_SW  * D;          // [N_PFAS,D]
    float* cnt_pg = agg_sp + (size_t)N_PFAS * D;
    float* cnt_gp = cnt_pg + N_GW;
    float* cnt_ps = cnt_gp + N_PFAS;
    float* cnt_sp = cnt_ps + N_SW;
    bf16_t* wb    = (bf16_t*)(cnt_sp + N_PFAS);          // 7 x 65536 bf16, 16B-aligned

    // 1) zero accumulators + counts
    size_t zn = (size_t)(N_GW + N_PFAS + N_SW + N_PFAS) * D
              + (size_t)(N_GW + N_PFAS + N_SW + N_PFAS);
    int n4 = (int)(zn / 4);
    zero4_kernel<<<(n4 + 255) / 256, 256, 0, stream>>>((float4*)d_ws, n4);

    // 2) weights -> bf16, transposed (Wr_gp + Wr_sp pre-summed: both multiply x_pfas)
    wconv_t_kernel<<<256, 256, 0, stream>>>(Wl_pg, nullptr, wb + 0 * 65536);
    wconv_t_kernel<<<256, 256, 0, stream>>>(Wr_pg, nullptr, wb + 1 * 65536);
    wconv_t_kernel<<<256, 256, 0, stream>>>(Wl_gp, nullptr, wb + 2 * 65536);
    wconv_t_kernel<<<256, 256, 0, stream>>>(Wl_sp, nullptr, wb + 3 * 65536);
    wconv_t_kernel<<<256, 256, 0, stream>>>(Wr_gp, Wr_sp,   wb + 4 * 65536);
    wconv_t_kernel<<<256, 256, 0, stream>>>(Wl_ps, nullptr, wb + 5 * 65536);
    wconv_t_kernel<<<256, 256, 0, stream>>>(Wr_ps, nullptr, wb + 6 * 65536);

    // 3) edge scatter-means (sum + count)
    int sb = (E + 3) / 4;
    scatter_kernel<<<sb, 256, 0, stream>>>(x_pfas, e_pg_src, e_pg_dst, agg_pg, cnt_pg, E);
    scatter_kernel<<<sb, 256, 0, stream>>>(x_gw,   e_gp_src, e_gp_dst, agg_gp, cnt_gp, E);
    scatter_kernel<<<sb, 256, 0, stream>>>(x_pfas, e_ps_src, e_ps_dst, agg_ps, cnt_ps, E);
    scatter_kernel<<<sb, 256, 0, stream>>>(x_sw,   e_sp_src, e_sp_dst, agg_sp, cnt_sp, E);

    // 4) fused WMMA GEMMs + relu + head
    gemm_kernel<<<(N_GW + MT - 1) / MT, 256, 0, stream>>>(
        agg_pg, x_gw, nullptr,
        wb + 0 * 65536, wb + 1 * 65536, nullptr,
        cnt_pg, nullptr, nullptr, 2,
        bl_pg, nullptr, W_gw, b_gw,
        outp, N_GW, 1);
    gemm_kernel<<<(N_PFAS + MT - 1) / MT, 256, 0, stream>>>(
        agg_gp, agg_sp, x_pfas,
        wb + 2 * 65536, wb + 3 * 65536, wb + 4 * 65536,
        cnt_gp, cnt_sp, nullptr, 3,
        bl_gp, bl_sp, W_sites, b_sites,
        outp + N_GW, N_PFAS, 1);
    gemm_kernel<<<(N_SW + MT - 1) / MT, 256, 0, stream>>>(
        agg_ps, x_sw, nullptr,
        wb + 5 * 65536, wb + 6 * 65536, nullptr,
        cnt_ps, nullptr, nullptr, 2,
        bl_ps, nullptr, nullptr, nullptr,
        outp + N_GW + N_PFAS, N_SW, 0);
}
